// SlotAttention_56676388438354
// MI455X (gfx1250) — matrix operand order repaired and checked by reading, compiled, not run on hardware
//
#include <hip/hip_runtime.h>
#include <hip/hip_bf16.h>

// ---------------------------------------------------------------------------
// Slot Attention for MI455X (gfx1250).
// All GEMMs: v_wmma_f32_16x16x32_bf16, bf16 operands resident in global
// (converted once per tensor), double-buffered LDS staging via
// global_load_async_to_lds_b128 (s_wait_asynccnt) when available.
// Shapes: B=32, S=256, HW=2304, NS=64, MH=1024, NIT=3.
// ---------------------------------------------------------------------------

typedef __attribute__((ext_vector_type(16))) __bf16 v16bf;
typedef __attribute__((ext_vector_type(8)))  float  v8f;
typedef __attribute__((ext_vector_type(4)))  int    v4i;

#define BSZ   32
#define SSZ   256
#define HW    2304
#define NSL   64
#define ROWS  (BSZ * HW)          // 73728

#define EPI_NONE       0
#define EPI_SCALE      1
#define EPI_BIAS       2
#define EPI_BIAS_RELU  3
#define EPI_BIAS_RES   4

#if defined(__has_builtin)
#if __has_builtin(__builtin_amdgcn_global_load_async_to_lds_b128)
#define HAS_ASYNC_LDS 1
#endif
#if __has_builtin(__builtin_amdgcn_s_wait_asynccnt)
#define HAS_WAIT_ASYNC 1
#endif
#endif

#define AS1 __attribute__((address_space(1)))
#define AS3 __attribute__((address_space(3)))

__device__ __forceinline__ unsigned short f2bf(float f) {
  unsigned u = __builtin_bit_cast(unsigned, f);
  u += 0x7FFFu + ((u >> 16) & 1u);              // round-to-nearest-even
  return (unsigned short)(u >> 16);
}

// 16-byte global(bf16) -> LDS copy; async DMA on gfx1250 when exposed.
__device__ __forceinline__ void copy16(const unsigned short* g, unsigned short* l) {
#if defined(HAS_ASYNC_LDS)
  __builtin_amdgcn_global_load_async_to_lds_b128(
      (AS1 v4i*)(unsigned long long)(uintptr_t)g,
      (AS3 v4i*)(unsigned)(uintptr_t)l, 0, 0);
#else
  *(uint4*)l = *(const uint4*)g;
#endif
}

__device__ __forceinline__ void async_wait0() {
#if defined(HAS_WAIT_ASYNC)
  __builtin_amdgcn_s_wait_asynccnt(0);
#elif defined(HAS_ASYNC_LDS)
  asm volatile("s_wait_asynccnt 0x0" ::: "memory");
#endif
}

union AFrag { v16bf v; unsigned u[8]; };

// ---------------------------------------------------------------------------
// C[M,N] = epi( A[M,K] * B[K,N] ),  A/B bf16, C f32 and/or bf16.
// aT==0: A bf16 row-major.  aT==1: A is f32 stored K-major (attn^T case).
// bT==0: B bf16 K-major.    bT==1: B bf16 N-major (q^T case).
// Tiles BM x BN, 8 waves of 32x32 (2x2 WMMA), LDS double-buffered.
// Requires M%BM==0, N%BN==0, K%32==0 (holds for all call sites).
// ---------------------------------------------------------------------------
template<int BM, int BN>
__global__ __launch_bounds__(256) void gemm_bf16_wmma(
    const void* __restrict__ A, long sA, int lda, int aT,
    const unsigned short* __restrict__ Bm, long sB, int ldb, int bT,
    float* __restrict__ C, unsigned short* __restrict__ Cb, long sC, int ldc,
    const float* __restrict__ bias, const float* __restrict__ resid, long sR,
    float scale, int K, int epi)
{
  constexpr int SA = 40;                 // LDS row stride (bf16): 80B = 5*16 ok
  constexpr int SB = BN + 8;             // 64->144B, 128->272B: both 16B mult
  constexpr int WAVES_M = BM / 32;
  constexpr int WAVES_N = 8 / WAVES_M;
  static_assert(WAVES_N * 32 == BN, "tile/wave mismatch");

  __shared__ unsigned short Asm[2][BM * SA];
  __shared__ unsigned short Bsm[2][32 * SB];

  const int tid = threadIdx.x;
  const int m0  = blockIdx.y * BM;
  const int n0  = blockIdx.x * BN;
  const long z  = blockIdx.z;

  const unsigned short* Abf = nullptr;
  const float*          Af  = nullptr;
  if (aT) Af  = (const float*)A + z * sA;
  else    Abf = (const unsigned short*)A + z * sA;
  Bm += z * sB;
  if (C)  C  += z * sC;
  if (Cb) Cb += z * sC;
  if (resid) resid += z * sR;

  const int wave = tid >> 5;
  const int lane = tid & 31;
  const int wm = (wave / WAVES_N) * 32;
  const int wn = (wave % WAVES_N) * 32;

  auto stage = [&](int buf, int k0) {
    unsigned short* As = Asm[buf];
    unsigned short* Bs = Bsm[buf];
    if (!aT) {                                     // pure bf16 16B copies
      #pragma unroll
      for (int it = 0; it < (BM * 32) / (8 * 256); ++it) {
        int idx8 = tid + it * 256;
        int m = idx8 >> 2;                         // (idx8*8)/32
        int k = (idx8 & 3) << 3;                   // (idx8*8)%32
        copy16(Abf + (long)(m0 + m) * lda + k0 + k, &As[m * SA + k]);
      }
    } else {                                       // f32 K-major, cvt (K=64)
      #pragma unroll
      for (int it = 0; it < (BM * 32) / 256; ++it) {
        int idx = tid + it * 256;
        int m  = idx % BM;                         // m fastest: coalesced
        int kk = idx / BM;
        As[m * SA + kk] = f2bf(Af[(long)(k0 + kk) * lda + (m0 + m)]);
      }
    }
    if (!bT) {                                     // bf16 K-major 16B copies
      #pragma unroll
      for (int it = 0; it < (32 * BN) / (8 * 256); ++it) {
        int idx8 = tid + it * 256;
        int nn = (idx8 << 3) % BN;
        int kk = (idx8 << 3) / BN;
        copy16(Bm + (long)(k0 + kk) * ldb + n0 + nn, &Bs[kk * SB + nn]);
      }
    } else {                                       // bf16 N-major, scalar copy
      #pragma unroll
      for (int it = 0; it < (32 * BN) / 256; ++it) {
        int idx = tid + it * 256;
        int kk = idx & 31;                         // k fastest: coalesced
        int nn = idx >> 5;
        Bs[kk * SB + nn] = Bm[(long)(n0 + nn) * ldb + k0 + kk];
      }
    }
  };

  v8f acc[2][2] = {};

  stage(0, 0);
  async_wait0();
  __syncthreads();

  for (int k0 = 0; k0 < K; k0 += 32) {
    const int buf = (k0 >> 5) & 1;
    const bool more = (k0 + 32) < K;
    if (more) stage(buf ^ 1, k0 + 32);             // overlap DMA with WMMA

    const int mr = lane & 15;
    const int kb = (lane >> 4) << 3;
    AFrag a[2], b[2];
    #pragma unroll
    for (int i = 0; i < 2; ++i) {
      const unsigned short* arow = &Asm[buf][(wm + i * 16 + mr) * SA];
      #pragma unroll
      for (int r = 0; r < 4; ++r) {
        a[i].u[r]     = *(const unsigned*)(arow + kb + 2 * r);
        a[i].u[4 + r] = *(const unsigned*)(arow + 16 + kb + 2 * r);
      }
    }
    #pragma unroll
    for (int j = 0; j < 2; ++j) {
      const unsigned short* bcol = &Bsm[buf][wn + j * 16 + mr];
      #pragma unroll
      for (int r = 0; r < 8; ++r) {
        const int kk = (r < 4) ? (kb + 2 * r) : (16 + kb + 2 * (r - 4));
        b[j].u[r] = (unsigned)bcol[kk * SB] | ((unsigned)bcol[(kk + 1) * SB] << 16);
      }
    }
    #pragma unroll
    for (int i = 0; i < 2; ++i)
      #pragma unroll
      for (int j = 0; j < 2; ++j)
        acc[i][j] = __builtin_amdgcn_wmma_f32_16x16x32_bf16(
            false, a[i].v, false, b[j].v, (short)0, acc[i][j], false, false);

    if (more) {
      async_wait0();
      __syncthreads();
    }
  }

  const int rb = m0 + wm + ((lane >> 4) << 3);
  const int cb = n0 + wn + (lane & 15);
  #pragma unroll
  for (int i = 0; i < 2; ++i) {
    #pragma unroll
    for (int j = 0; j < 2; ++j) {
      const int col = cb + j * 16;
      const float bv = (epi >= EPI_BIAS) ? bias[col] : 0.f;
      #pragma unroll
      for (int r = 0; r < 8; ++r) {
        const int row = rb + i * 16 + r;
        float v = acc[i][j][r];
        if (epi == EPI_SCALE)          v *= scale;
        else if (epi == EPI_BIAS)      v += bv;
        else if (epi == EPI_BIAS_RELU) v = fmaxf(v + bv, 0.f);
        else if (epi == EPI_BIAS_RES)  v += bv + resid[(long)row * ldc + col];
        if (C)  C[(long)row * ldc + col]  = v;
        if (Cb) Cb[(long)row * ldc + col] = f2bf(v);
      }
    }
  }
}

// ---- f32 -> bf16 converters (weights once per call, activations) ----
__global__ __launch_bounds__(256) void cvt_bf16(const float* __restrict__ s,
                                                unsigned short* __restrict__ d) {
  int i = blockIdx.x * 256 + threadIdx.x;
  d[i] = f2bf(s[i]);
}
// dst[c*rows + r] = src[r*cols + c]   (builds K-major W^T copies)
__global__ __launch_bounds__(256) void cvt_trans_bf16(const float* __restrict__ s,
                                                      unsigned short* __restrict__ d,
                                                      int rows, int cols) {
  int i = blockIdx.x * 256 + threadIdx.x;
  int r = i / cols, c = i % cols;
  d[(long)c * rows + r] = f2bf(s[i]);
}

// ---- softmax over HW per (b, slot) row, then +EPS ----
__global__ __launch_bounds__(256) void softmax_rows(float* __restrict__ attn) {
  float* p = attn + (long)blockIdx.x * HW;
  const int tid = threadIdx.x;
  __shared__ float red[256];
  float m = -3.4e38f;
  for (int t = tid; t < HW; t += 256) m = fmaxf(m, p[t]);
  red[tid] = m; __syncthreads();
  for (int s = 128; s > 0; s >>= 1) {
    if (tid < s) red[tid] = fmaxf(red[tid], red[tid + s]);
    __syncthreads();
  }
  m = red[0]; __syncthreads();
  float sum = 0.f;
  for (int t = tid; t < HW; t += 256) {
    float e = __expf(p[t] - m);
    p[t] = e; sum += e;
  }
  red[tid] = sum; __syncthreads();
  for (int s = 128; s > 0; s >>= 1) {
    if (tid < s) red[tid] += red[tid + s];
    __syncthreads();
  }
  const float inv = 1.f / red[0];
  for (int t = tid; t < HW; t += 256) p[t] = p[t] * inv + 1e-8f;
}

// ---- normalize over the NS axis ----
__global__ __launch_bounds__(256) void colnorm(float* __restrict__ attn) {
  int idx = blockIdx.x * 256 + threadIdx.x;      // < B*HW
  int b = idx / HW, j = idx % HW;
  float* p = attn + (long)b * NSL * HW + j;
  float s = 0.f;
  #pragma unroll 4
  for (int i = 0; i < NSL; ++i) s += p[(long)i * HW];
  const float inv = 1.f / s;
  #pragma unroll 4
  for (int i = 0; i < NSL; ++i) p[(long)i * HW] *= inv;
}

// ---- GRUCell pointwise; writes f32 state + bf16 copy for next GEMMs ----
__global__ __launch_bounds__(256) void gru_kernel(
    const float* __restrict__ gi, const float* __restrict__ gh,
    float* __restrict__ h, unsigned short* __restrict__ hb)
{
  int idx = blockIdx.x * 256 + threadIdx.x;      // < ROWS*SSZ
  int row = idx >> 8;
  int c = idx & 255;
  long base = (long)row * 768 + c;
  float ir = gi[base], iz = gi[base + 256], in = gi[base + 512];
  float hr = gh[base], hz = gh[base + 256], hn = gh[base + 512];
  float hv = h[idx];
  float r = 1.f / (1.f + __expf(-(ir + hr)));
  float z = 1.f / (1.f + __expf(-(iz + hz)));
  float n = tanhf(in + r * hn);
  float out = (1.f - z) * n + z * hv;
  h[idx] = out;
  hb[idx] = f2bf(out);
}

template<int BM, int BN>
static inline void gemm(hipStream_t st,
                        const void* A, long sA, int lda, int aT,
                        const unsigned short* B, long sB, int ldb, int bT,
                        float* C, unsigned short* Cb, long sC, int ldc,
                        const float* bias, const float* resid, long sR,
                        float scale, int M, int N, int K, int batch, int epi)
{
  dim3 g(N / BN, M / BM, batch), blk(256);
  hipLaunchKernelGGL((gemm_bf16_wmma<BM, BN>), g, blk, 0, st,
                     A, sA, lda, aT, B, sB, ldb, bT, C, Cb, sC, ldc,
                     bias, resid, sR, scale, K, epi);
}

extern "C" void kernel_launch(void* const* d_in, const int* in_sizes, int n_in,
                              void* d_out, int out_size, void* d_ws, size_t ws_size,
                              hipStream_t stream) {
  const float* inp  = (const float*)d_in[0];
  const float* pro  = (const float*)d_in[1];
  const float* Wk   = (const float*)d_in[2];
  const float* Wq   = (const float*)d_in[3];
  const float* m1W1 = (const float*)d_in[4];
  const float* m1b1 = (const float*)d_in[5];
  const float* m1W2 = (const float*)d_in[6];
  const float* m1b2 = (const float*)d_in[7];
  const float* m2W1 = (const float*)d_in[8];
  const float* m2b1 = (const float*)d_in[9];
  const float* m2W2 = (const float*)d_in[10];
  const float* m2b2 = (const float*)d_in[11];
  const float* gWih = (const float*)d_in[12];
  const float* gWhh = (const float*)d_in[13];
  const float* gbih = (const float*)d_in[14];
  const float* gbhh = (const float*)d_in[15];

  char* base = (char*)d_ws;
  size_t off = 0;
  auto takeB = [&](size_t bytes) {
    void* p = base + off;
    off = (off + bytes + 255) & ~(size_t)255;
    return p;
  };
  typedef unsigned short us;
  // bf16 weight copies (K-major; GRU weights transposed into K-major)
  us* wq_b   = (us*)takeB(65536 * 2);
  us* m1w1_b = (us*)takeB(65536 * 2);
  us* m1w2_b = (us*)takeB(65536 * 2);
  us* wk_b   = (us*)takeB(65536 * 2);
  us* m2w1_b = (us*)takeB(262144 * 2);
  us* m2w2_b = (us*)takeB(262144 * 2);
  us* wihT_b = (us*)takeB(196608 * 2);
  us* whhT_b = (us*)takeB(196608 * 2);
  // small activations
  us* pro_b  = (us*)takeB(524288 * 2);
  us* t1_b   = (us*)takeB(524288 * 2);
  us* pro2_b = (us*)takeB(524288 * 2);
  us* kv_b   = (us*)takeB(524288 * 2);
  // big buffers
  float* slots   = (float*)takeB((size_t)ROWS * 256 * 4);
  us*    slots_b = (us*)takeB((size_t)ROWS * 256 * 2);
  us*    qu_b    = (us*)takeB((size_t)ROWS * 256 * 2);   // q, then updates
  float* attn    = (float*)takeB((size_t)BSZ * NSL * HW * 4);
  float* gi      = (float*)takeB((size_t)ROWS * 768 * 4);
  float* gh      = (float*)takeB((size_t)ROWS * 768 * 4);
  us*    h1_b    = (us*)gi;   // mlp2 hidden (ROWS*1024 bf16) reuses gi+gh

  auto cvt = [&](const float* s, us* d, long n) {
    hipLaunchKernelGGL(cvt_bf16, dim3(n / 256), dim3(256), 0, stream, s, d);
  };
  cvt(Wq, wq_b, 65536);       cvt(m1W1, m1w1_b, 65536);
  cvt(m1W2, m1w2_b, 65536);   cvt(Wk, wk_b, 65536);
  cvt(m2W1, m2w1_b, 262144);  cvt(m2W2, m2w2_b, 262144);
  cvt(pro, pro_b, 524288);
  hipLaunchKernelGGL(cvt_trans_bf16, dim3(196608 / 256), dim3(256), 0, stream,
                     gWih, wihT_b, 768, 256);
  hipLaunchKernelGGL(cvt_trans_bf16, dim3(196608 / 256), dim3(256), 0, stream,
                     gWhh, whhT_b, 768, 256);
  // slots: f32 raw view of inputs + bf16 copy
  (void)hipMemcpyAsync(slots, inp, (size_t)ROWS * 256 * 4,
                       hipMemcpyDeviceToDevice, stream);
  cvt(inp, slots_b, (long)ROWS * 256);

  // mlp1 on prototypes, then k (== v)
  gemm<128,64>(stream, pro_b, 0, 256, 0, m1w1_b, 0, 256, 0,
               nullptr, t1_b, 0, 256, m1b1, nullptr, 0, 1.f,
               2048, 256, 256, 1, EPI_BIAS_RELU);
  gemm<128,64>(stream, t1_b, 0, 256, 0, m1w2_b, 0, 256, 0,
               nullptr, pro2_b, 0, 256, m1b2, nullptr, 0, 1.f,
               2048, 256, 256, 1, EPI_BIAS);
  gemm<128,64>(stream, pro2_b, 0, 256, 0, wk_b, 0, 256, 0,
               nullptr, kv_b, 0, 256, nullptr, nullptr, 0, 1.f,
               2048, 256, 256, 1, EPI_NONE);

  for (int it = 0; it < 3; ++it) {
    // q = (slots @ Wq) * S^-0.5   (bf16 only)
    gemm<128,64>(stream, slots_b, 0, 256, 0, wq_b, 0, 256, 0,
                 nullptr, qu_b, 0, 256, nullptr, nullptr, 0, 0.0625f,
                 ROWS, 256, 256, 1, EPI_SCALE);
    // logits[b] = k[b] @ q[b]^T -> (64 x 2304) f32, batched
    gemm<64,128>(stream, kv_b, (long)NSL * 256, 256, 0,
                 qu_b, (long)HW * 256, 256, 1,
                 attn, nullptr, (long)NSL * HW, HW,
                 nullptr, nullptr, 0, 1.f, NSL, HW, 256, BSZ, EPI_NONE);
    hipLaunchKernelGGL(softmax_rows, dim3(BSZ * NSL), dim3(256), 0, stream, attn);
    hipLaunchKernelGGL(colnorm, dim3((BSZ * HW) / 256), dim3(256), 0, stream, attn);
    // updates[b] = attn[b]^T @ v[b]  (bf16 out, overwrites q buffer)
    gemm<128,64>(stream, attn, (long)NSL * HW, HW, 1,
                 kv_b, (long)NSL * 256, 256, 0,
                 nullptr, qu_b, (long)HW * 256, 256,
                 nullptr, nullptr, 0, 1.f, HW, 256, NSL, BSZ, EPI_NONE);
    // GRU gates (f32): gi = upd @ Wih^T + bih ; gh = slots @ Whh^T + bhh
    gemm<128,64>(stream, qu_b, 0, 256, 0, wihT_b, 0, 768, 0,
                 gi, nullptr, 0, 768, gbih, nullptr, 0, 1.f,
                 ROWS, 768, 256, 1, EPI_BIAS);
    gemm<128,64>(stream, slots_b, 0, 256, 0, whhT_b, 0, 768, 0,
                 gh, nullptr, 0, 768, gbhh, nullptr, 0, 1.f,
                 ROWS, 768, 256, 1, EPI_BIAS);
    hipLaunchKernelGGL(gru_kernel, dim3((ROWS * 256) / 256), dim3(256), 0,
                       stream, gi, gh, slots, slots_b);
    // mlp2 residual: slots += relu(slots @ m2W1 + b1) @ m2W2 + b2
    gemm<128,64>(stream, slots_b, 0, 256, 0, m2w1_b, 0, 1024, 0,
                 nullptr, h1_b, 0, 1024, m2b1, nullptr, 0, 1.f,
                 ROWS, 1024, 256, 1, EPI_BIAS_RELU);
    gemm<128,64>(stream, h1_b, 0, 1024, 0, m2w2_b, 0, 256, 0,
                 slots, slots_b, 0, 256, m2b2, slots, 0, 1.f,
                 ROWS, 256, 1024, 1, EPI_BIAS_RES);
  }

  (void)hipMemcpyAsync(d_out, slots, (size_t)ROWS * 256 * 4,
                       hipMemcpyDeviceToDevice, stream);
}
